// Encoder_80032420593996
// MI455X (gfx1250) — matrix operand order repaired and checked by reading, not compiled, not run on hardware
//
#include <hip/hip_runtime.h>
#include <math.h>

typedef __attribute__((ext_vector_type(16))) __bf16 v16bf;
typedef __attribute__((ext_vector_type(8)))  float  v8f;

#define NEG_SLOPE 0.2f

// ---------------------------------------------------------------------------
// float -> bf16, round-to-nearest-even (pure integer ops: guaranteed codegen)
// ---------------------------------------------------------------------------
__device__ __forceinline__ __bf16 f2bf(float f) {
    unsigned u = __float_as_uint(f);
    u += 0x7FFFu + ((u >> 16) & 1u);
    unsigned short h = (unsigned short)(u >> 16);
    return __builtin_bit_cast(__bf16, h);
}

// ---------------------------------------------------------------------------
// Pack W[G x Hdim] (fp32, row-major) into per-lane B-fragment layout (bf16):
//   Bpack[ks][colT][lane][16]   (16 bf16 = 32 B per lane, zero-padded K tail)
// B layout (16-bit B 32x16): lane = half*16 + n ; VGPR j holds K = (j<4?0:16)+8*half+2*(j&3),
// (j<4?0:16)+8*half+2*(j&3)+1 at column n.  One wave per (ks, colT) tile.
// ---------------------------------------------------------------------------
__global__ __launch_bounds__(32) void pack_b_kernel(
    const float* __restrict__ W, __bf16* __restrict__ Bpack,
    int G, int Hdim, int K32)
{
    const int ks   = blockIdx.x;
    const int colT = blockIdx.y;
    const int lane = threadIdx.x & 31;
    const int half = lane >> 4;
    const int r    = lane & 15;
    const int col  = colT * 16 + r;
    const int k    = ks * 32;

    v16bf v;
#pragma unroll
    for (int j = 0; j < 8; ++j) {
        const int kk = k + ((j < 4) ? 0 : 16) + half * 8 + (j & 3) * 2;
        const float b0 = (kk     < G) ? W[(size_t)kk * Hdim + col]       : 0.f;
        const float b1 = (kk + 1 < G) ? W[(size_t)(kk + 1) * Hdim + col] : 0.f;
        v[2 * j]     = f2bf(b0);
        v[2 * j + 1] = f2bf(b1);
    }
    __bf16* dst = Bpack + (((size_t)ks * (Hdim >> 4) + colT) * 32 + lane) * 16;
    *(v16bf*)dst = v;
}

// ---------------------------------------------------------------------------
// GEMM: Hout[N x Hdim] = X[N x G] @ W[G x Hdim] via v_wmma_f32_16x16x32_bf16.
//
// Block = 4 waves, one 16-row x 256-col output tile per block. Each wave
// accumulates a strided quarter of the K-steps (X still read exactly once
// from HBM), then partials are reduced through LDS. 4x the wave count of the
// single-wave version -> ~5 waves/SIMD to hide B-fragment L2 latency behind
// other waves' WMMAs.
// ---------------------------------------------------------------------------
#define COL_TILES 16   // Hdim / 16, fixed for this problem (H = 256)
#define KWAVES    4    // K-split factor (waves per block)

__global__ __launch_bounds__(32 * KWAVES) void gemm_bf16_wmma(
    const float* __restrict__ X, const __bf16* __restrict__ Bpack,
    float* __restrict__ Hout, int N, int G, int Hdim, int K32)
{
    const int lane = threadIdx.x & 31;
    const int w    = threadIdx.x >> 5;   // wave id in block (0..3)
    const int half = lane >> 4;
    const int r    = lane & 15;
    const int rowT = blockIdx.x;

    // partial accumulators: part[w][t][lane][8] floats
    __shared__ float part[KWAVES * COL_TILES * 32 * 8];

    const float* __restrict__ Xrow = X + (size_t)(rowT * 16 + r) * (size_t)G;

    v8f acc[COL_TILES];
#pragma unroll
    for (int t = 0; t < COL_TILES; ++t) {
        v8f z = {0.f, 0.f, 0.f, 0.f, 0.f, 0.f, 0.f, 0.f};
        acc[t] = z;
    }

    const size_t bStep = (size_t)COL_TILES * 32 * 16;  // Bpack elems per k-step

    for (int ks = w; ks < K32; ks += KWAVES) {
        const int k = ks * 32;
        const __bf16* bpk = Bpack + (size_t)ks * bStep + (size_t)lane * 16;

        // A fragment: lane = row r; K pairs at half*8 + {0..7} and 16+half*8+{0..7}
        v16bf a;
        if (k + 32 <= G) {
            const float4 f0 = *(const float4*)(Xrow + k + half * 8);
            const float4 f1 = *(const float4*)(Xrow + k + half * 8 + 4);
            const float4 f2 = *(const float4*)(Xrow + k + 16 + half * 8);
            const float4 f3 = *(const float4*)(Xrow + k + 16 + half * 8 + 4);
            a[0]  = f2bf(f0.x); a[1]  = f2bf(f0.y); a[2]  = f2bf(f0.z); a[3]  = f2bf(f0.w);
            a[4]  = f2bf(f1.x); a[5]  = f2bf(f1.y); a[6]  = f2bf(f1.z); a[7]  = f2bf(f1.w);
            a[8]  = f2bf(f2.x); a[9]  = f2bf(f2.y); a[10] = f2bf(f2.z); a[11] = f2bf(f2.w);
            a[12] = f2bf(f3.x); a[13] = f2bf(f3.y); a[14] = f2bf(f3.z); a[15] = f2bf(f3.w);
        } else {
            // K tail (Bpack is zero-padded; guard A element-wise)
#pragma unroll
            for (int j = 0; j < 8; ++j) {
                const int off = k + ((j < 4) ? 0 : 16) + half * 8 + (j & 3) * 2;
                const float ax = (off     < G) ? Xrow[off]     : 0.f;
                const float ay = (off + 1 < G) ? Xrow[off + 1] : 0.f;
                a[2 * j]     = f2bf(ax);
                a[2 * j + 1] = f2bf(ay);
            }
        }

#pragma unroll
        for (int t = 0; t < COL_TILES; ++t) {
            const v16bf b = *(const v16bf*)(bpk + (size_t)t * 512);  // 32 lanes * 16
            acc[t] = __builtin_amdgcn_wmma_f32_16x16x32_bf16(
                false, a, false, b, (short)0, acc[t], false, false);
        }
    }

    // stash partials in LDS: part[w][t][lane][8]
#pragma unroll
    for (int t = 0; t < COL_TILES; ++t) {
        *(v8f*)&part[(((size_t)w * COL_TILES + t) * 32 + lane) * 8] = acc[t];
    }
    __syncthreads();

    // each wave reduces KWAVES partials for its 4 column tiles and stores
#pragma unroll
    for (int tt = 0; tt < COL_TILES / KWAVES; ++tt) {
        const int t = w * (COL_TILES / KWAVES) + tt;
        v8f s = *(const v8f*)&part[(((size_t)0 * COL_TILES + t) * 32 + lane) * 8];
#pragma unroll
        for (int sw = 1; sw < KWAVES; ++sw) {
            s = s + *(const v8f*)&part[(((size_t)sw * COL_TILES + t) * 32 + lane) * 8];
        }
        // C/D layout: value i -> row 8*half+i, lane r -> col r
        const int col = t * 16 + r;
#pragma unroll
        for (int i = 0; i < 8; ++i) {
            Hout[(size_t)(rowT * 16 + half * 8 + i) * Hdim + col] = s[i];
        }
    }
}

// ---------------------------------------------------------------------------
// a[row] = dot(Hm[row,:], att)   (one wave per row)
// ---------------------------------------------------------------------------
__global__ __launch_bounds__(256) void rowdot_kernel(
    const float* __restrict__ Hm, const float* __restrict__ att,
    float* __restrict__ out, int N, int Hdim)
{
    const int wave = (int)((blockIdx.x * blockDim.x + threadIdx.x) >> 5);
    const int lane = threadIdx.x & 31;
    if (wave >= N) return;
    const float* __restrict__ row = Hm + (size_t)wave * Hdim;
    float s = 0.f;
    for (int h = lane; h < Hdim; h += 32) s += row[h] * att[h];
#pragma unroll
    for (int o = 16; o > 0; o >>= 1) s += __shfl_xor(s, o, 32);
    if (lane == 0) out[wave] = s;
}

// ---------------------------------------------------------------------------
// init out / segment buffers
// ---------------------------------------------------------------------------
__global__ void init_kernel(float* __restrict__ out, float* __restrict__ m,
                            float* __restrict__ den, int N1, int total)
{
    const int i = blockIdx.x * blockDim.x + threadIdx.x;
    if (i < N1) { m[i] = -INFINITY; den[i] = 0.f; }
    if (i < total) out[i] = 0.f;
}

__device__ __forceinline__ float leaky(float x) {
    return x > 0.f ? x : NEG_SLOPE * x;
}

__device__ __forceinline__ void atomicMaxF(float* addr, float val) {
    if (val >= 0.f) atomicMax((int*)addr, __float_as_int(val));
    else            atomicMin((unsigned int*)addr, __float_as_uint(val));
}

// ---------------------------------------------------------------------------
// segment max of edge scores over dst
// ---------------------------------------------------------------------------
__global__ void edge_max_kernel(const int* __restrict__ src, const int* __restrict__ dst,
                                const float* __restrict__ a_s, const float* __restrict__ a_d,
                                float* __restrict__ m, int E)
{
    const int i = blockIdx.x * blockDim.x + threadIdx.x;
    if (i >= E) return;
    const int s = src[i], d = dst[i];
    atomicMaxF(&m[d], leaky(a_s[s] + a_d[d]));
}

// ---------------------------------------------------------------------------
// denom[d] = sum exp(e - m[d])
// ---------------------------------------------------------------------------
__global__ void edge_sum_kernel(const int* __restrict__ src, const int* __restrict__ dst,
                                const float* __restrict__ a_s, const float* __restrict__ a_d,
                                const float* __restrict__ m, float* __restrict__ den, int E)
{
    const int i = blockIdx.x * blockDim.x + threadIdx.x;
    if (i >= E) return;
    const int s = src[i], d = dst[i];
    const float e = leaky(a_s[s] + a_d[d]);
    atomicAdd(&den[d], __expf(e - m[d]));
}

// ---------------------------------------------------------------------------
// out[d,:] += alpha * h_src[s,:]   (one wave per edge)
// ---------------------------------------------------------------------------
__global__ __launch_bounds__(256) void scatter_kernel(
    const int* __restrict__ src, const int* __restrict__ dst,
    const float* __restrict__ a_s, const float* __restrict__ a_d,
    const float* __restrict__ m, const float* __restrict__ den,
    const float* __restrict__ h_src, float* __restrict__ out, int E, int Hdim)
{
    const int e    = (int)((blockIdx.x * blockDim.x + threadIdx.x) >> 5);
    const int lane = threadIdx.x & 31;
    if (e >= E) return;
    const int s = src[e], d = dst[e];
    const float sc    = leaky(a_s[s] + a_d[d]);
    const float alpha = __expf(sc - m[d]) / den[d];
    const float* __restrict__ hr = h_src + (size_t)s * Hdim;
    float* __restrict__ orow = out + (size_t)d * Hdim;
    for (int h = lane; h < Hdim; h += 32)
        atomicAdd(&orow[h], hr[h] * alpha);
}

// ---------------------------------------------------------------------------
// out = relu(out + bias)
// ---------------------------------------------------------------------------
__global__ void finalize_kernel(float* __restrict__ out, const float* __restrict__ bias,
                                int total, int Hdim)
{
    const int i = blockIdx.x * blockDim.x + threadIdx.x;
    if (i >= total) return;
    const float v = out[i] + bias[i % Hdim];
    out[i] = v > 0.f ? v : 0.f;
}

// ---------------------------------------------------------------------------
extern "C" void kernel_launch(void* const* d_in, const int* in_sizes, int n_in,
                              void* d_out, int out_size, void* d_ws, size_t ws_size,
                              hipStream_t stream)
{
    const int*   pi   = (const int*)  d_in[0];  // [2, E]
    const float* X1   = (const float*)d_in[1];  // [N1, G1]
    const float* X2   = (const float*)d_in[2];  // [N2, G2]
    const float* Wsrc = (const float*)d_in[3];  // [G2, H]
    const float* Wdst = (const float*)d_in[4];  // [G1, H]
    const float* attS = (const float*)d_in[5];  // [H]
    const float* attD = (const float*)d_in[6];  // [H]
    const float* bias = (const float*)d_in[7];  // [H]
    float* out = (float*)d_out;                 // [N1, H]

    const int E  = in_sizes[0] / 2;
    const int H  = in_sizes[5];
    const int G2 = in_sizes[3] / H;
    const int G1 = in_sizes[4] / H;
    const int N2 = in_sizes[2] / G2;
    const int N1 = in_sizes[1] / G1;

    const int K32s = (G2 + 31) / 32;   // k-steps for W_src
    const int K32d = (G1 + 31) / 32;   // k-steps for W_dst

    const int* src = pi;       // row 0
    const int* dst = pi + E;   // row 1

    // workspace layout (floats), Bpack (bf16) reused for both weight matrices
    float* h_src = (float*)d_ws;                  // N2*H
    float* h_dst = h_src + (size_t)N2 * H;        // N1*H
    float* a_s   = h_dst + (size_t)N1 * H;        // N2
    float* a_d   = a_s + N2;                      // N1
    float* m     = a_d + N1;                      // N1
    float* den   = m + N1;                        // N1
    __bf16* Bpack = (__bf16*)(den + N1);          // max(K32) * (H/16) * 32 * 16

    const int total = N1 * H;

    // init out / m / den
    init_kernel<<<(total + 255) / 256, 256, 0, stream>>>(out, m, den, N1, total);

    // h_src = X2 @ W_src
    pack_b_kernel<<<dim3(K32s, H / 16), 32, 0, stream>>>(Wsrc, Bpack, G2, H, K32s);
    gemm_bf16_wmma<<<N2 / 16, 32 * KWAVES, 0, stream>>>(X2, Bpack, h_src, N2, G2, H, K32s);

    // h_dst = X1 @ W_dst (Bpack reused)
    pack_b_kernel<<<dim3(K32d, H / 16), 32, 0, stream>>>(Wdst, Bpack, G1, H, K32d);
    gemm_bf16_wmma<<<N1 / 16, 32 * KWAVES, 0, stream>>>(X1, Bpack, h_dst, N1, G1, H, K32d);

    // attention row-dots (one wave per row)
    rowdot_kernel<<<(N2 + 7) / 8, 256, 0, stream>>>(h_src, attS, a_s, N2, H);
    rowdot_kernel<<<(N1 + 7) / 8, 256, 0, stream>>>(h_dst, attD, a_d, N1, H);

    // edge softmax (segment max, segment sum) + scatter aggregation
    edge_max_kernel<<<(E + 255) / 256, 256, 0, stream>>>(src, dst, a_s, a_d, m, E);
    edge_sum_kernel<<<(E + 255) / 256, 256, 0, stream>>>(src, dst, a_s, a_d, m, den, E);
    scatter_kernel<<<(E + 7) / 8, 256, 0, stream>>>(src, dst, a_s, a_d, m, den,
                                                    h_src, out, E, H);

    // bias + relu
    finalize_kernel<<<(total + 255) / 256, 256, 0, stream>>>(out, bias, total, H);
}
